// GPTQLinear_67224828117444
// MI455X (gfx1250) — matrix operand order repaired and checked by reading, not compiled, not run on hardware
//
#include <hip/hip_runtime.h>

typedef __attribute__((ext_vector_type(16))) _Float16 v16h;
typedef __attribute__((ext_vector_type(8)))  _Float16 v8h;
typedef __attribute__((ext_vector_type(4)))  _Float16 v4h;
typedef __attribute__((ext_vector_type(8)))  float    v8f;
typedef __attribute__((ext_vector_type(4)))  float    v4f;

namespace {
constexpr int kInF  = 4096;
constexpr int kOutF = 4096;
constexpr int kGS   = 128;
constexpr int kTok  = 4 * 2048;          // 8192 tokens

constexpr int BM  = 128;                 // tokens per block tile
constexpr int BN  = 128;                 // out-features per block tile
constexpr int BK  = 32;                  // one v_wmma_f32_16x16x32_f16 K step
constexpr int LDA = BK + 8;              // padded LDS row stride (halfs) -> no 4-way bank conflicts
}

// ---------------------------------------------------------------------------
// Pass 1: dequantize groupwise int4 and scatter through the activation
// permutation:  W16[o, perm[j]] = ((nibble - 8) * scale[o, j/128])  as f16.
// One thread per packed byte (two int4 values, always in the same scale group).
// Every element of W16 is written exactly once per call (perm is a bijection),
// so the workspace is fully re-initialized each launch (deterministic).
// ---------------------------------------------------------------------------
__global__ __launch_bounds__(256)
void gptq_dequant_permute_w(const int* __restrict__ w_packed,
                            const float* __restrict__ w_scales,
                            const int* __restrict__ perm,
                            _Float16* __restrict__ w16) {
  const int t  = blockIdx.x * 256 + threadIdx.x;   // 0 .. OUT_F*IN_F/2 - 1
  const int o  = t >> 11;                          // / (kInF/2)
  const int jb = t & 2047;                         // % (kInF/2)
  const int byte = w_packed[t] & 0xFF;
  const int j0   = jb << 1;
  const float scale = w_scales[o * (kInF / kGS) + (j0 >> 7)];
  const int lo = (byte & 0xF) - 8;
  const int hi = ((byte >> 4) & 0xF) - 8;
  const int p0 = perm[j0];
  const int p1 = perm[j0 + 1];
  w16[(size_t)o * kInF + p0] = (_Float16)((float)lo * scale);
  w16[(size_t)o * kInF + p1] = (_Float16)((float)hi * scale);
}

// ---------------------------------------------------------------------------
// Pass 2: f16 WMMA GEMM  out[M,N] = x[M,K] * W16[N,K]^T + bias[N]
// 256 threads = 8 waves (wave32). Block tile 128x128, wave tile 32x64
// (2x4 WMMA 16x16 tiles), K stepped by 32 through LDS-staged operands.
// ---------------------------------------------------------------------------
__global__ __launch_bounds__(256)
void gptq_gemm_wmma(const float* __restrict__ x,
                    const _Float16* __restrict__ w16,
                    const float* __restrict__ bias,
                    float* __restrict__ out) {
  __shared__ _Float16 Ah[BM * LDA];   // [token][k]  10 KB
  __shared__ _Float16 Bh[BN * LDA];   // [outf ][k]  10 KB

  const int tid  = threadIdx.x;
  const int lane = tid & 31;
  const int wave = tid >> 5;          // 0..7
  const int wm   = wave & 3;          // 4 wave-rows  * 32 tokens
  const int wn   = wave >> 2;         // 2 wave-cols  * 64 outf
  const int half = lane >> 4;         // ISA lane-half selector
  const int l16  = lane & 15;

  const int block_m = blockIdx.y * BM;
  const int block_n = blockIdx.x * BN;

  v8f acc[2][4];
#pragma unroll
  for (int mt = 0; mt < 2; ++mt)
#pragma unroll
    for (int nt = 0; nt < 4; ++nt) acc[mt][nt] = (v8f)0.0f;

  for (int k0 = 0; k0 < kInF; k0 += BK) {
    // --- stage A: 128x32 fp32 -> f16 (1024 float4's, 4 per thread) ---------
#pragma unroll
    for (int i = 0; i < 4; ++i) {
      const int v   = tid + i * 256;
      const int row = v >> 3;                 // /8 float4's per row
      const int kk  = (v & 7) << 2;
      const v4f xv = *(const v4f*)(x + (size_t)(block_m + row) * kInF + k0 + kk);
      *(v4h*)(Ah + row * LDA + kk) = __builtin_convertvector(xv, v4h);
    }
    // --- stage B: 128x32 f16 (512 16B vectors, 2 per thread) ---------------
#pragma unroll
    for (int i = 0; i < 2; ++i) {
      const int v   = tid + i * 256;
      const int row = v >> 2;                 // /4 v8h's per row
      const int kk  = (v & 3) << 3;
      *(v8h*)(Bh + row * LDA + kk) =
          *(const v8h*)(w16 + (size_t)(block_n + row) * kInF + k0 + kk);
    }
    // prefetch next K tile while we compute on this one (global_prefetch_b8)
    if (k0 + BK < kInF) {
      __builtin_prefetch(x + (size_t)(block_m + (tid >> 1)) * kInF + k0 + BK, 0, 1);
      __builtin_prefetch(w16 + (size_t)(block_n + (tid >> 1)) * kInF + k0 + BK, 0, 1);
    }
    __syncthreads();

    // --- A fragments: lane holds row (l16), K halves per ISA 7.12.2 --------
    v16h afrag[2];
#pragma unroll
    for (int mt = 0; mt < 2; ++mt) {
      const _Float16* ap = Ah + (wm * 32 + mt * 16 + l16) * LDA;
      const v8h lo = *(const v8h*)(ap + half * 8);        // K = h*8 .. h*8+7
      const v8h hi = *(const v8h*)(ap + 16 + half * 8);   // K = 16+h*8 ..
      afrag[mt] = __builtin_shufflevector(lo, hi, 0, 1, 2, 3, 4, 5, 6, 7,
                                          8, 9, 10, 11, 12, 13, 14, 15);
    }
    // --- B fragments: lane holds column (l16), K halves per ISA 7.12.2 -----
    v16h bfrag[4];
#pragma unroll
    for (int nt = 0; nt < 4; ++nt) {
      const _Float16* bp = Bh + (wn * 64 + nt * 16 + l16) * LDA + half * 8;
      const v8h lo = *(const v8h*)(bp);
      const v8h hi = *(const v8h*)(bp + 16);
      bfrag[nt] = __builtin_shufflevector(lo, hi, 0, 1, 2, 3, 4, 5, 6, 7,
                                          8, 9, 10, 11, 12, 13, 14, 15);
    }
    // --- 8 WMMAs per K step -----------------------------------------------
#pragma unroll
    for (int mt = 0; mt < 2; ++mt)
#pragma unroll
      for (int nt = 0; nt < 4; ++nt)
        acc[mt][nt] = __builtin_amdgcn_wmma_f32_16x16x32_f16(
            /*neg_a=*/false, afrag[mt], /*neg_b=*/false, bfrag[nt],
            /*c_mod=*/(short)0, acc[mt][nt],
            /*reuse_a=*/false, /*reuse_b=*/false);

    __syncthreads();
  }

  // --- epilogue: bias add + store (C layout: M = r + 8*half, N = l16) ------
#pragma unroll
  for (int nt = 0; nt < 4; ++nt) {
    const int gcol = block_n + wn * 64 + nt * 16 + l16;
    const float bv = bias[gcol];
#pragma unroll
    for (int mt = 0; mt < 2; ++mt) {
#pragma unroll
      for (int r = 0; r < 8; ++r) {
        const int grow = block_m + wm * 32 + mt * 16 + r + half * 8;
        out[(size_t)grow * kOutF + gcol] = acc[mt][nt][r] + bv;
      }
    }
  }
}

// ---------------------------------------------------------------------------
extern "C" void kernel_launch(void* const* d_in, const int* in_sizes, int n_in,
                              void* d_out, int out_size, void* d_ws, size_t ws_size,
                              hipStream_t stream) {
  const float* x        = (const float*)d_in[0];   // (4,2048,4096) fp32
  const int*   w_packed = (const int*)d_in[1];     // (4096,2048) int32 (1 byte used)
  const float* w_scales = (const float*)d_in[2];   // (4096,32) fp32
  const int*   perm     = (const int*)d_in[3];     // (4096,) int32
  const float* bias     = (const float*)d_in[4];   // (4096,) fp32
  float*       out      = (float*)d_out;           // (4,2048,4096) fp32

  _Float16* w16 = (_Float16*)d_ws;                 // 4096*4096 f16 = 32 MB scratch

  // Pass 1: dequant + act-order scatter into f16 weights
  const int n_bytes = kOutF * (kInF / 2);
  gptq_dequant_permute_w<<<n_bytes / 256, 256, 0, stream>>>(w_packed, w_scales,
                                                            perm, w16);
  // Pass 2: WMMA GEMM + bias
  dim3 grid(kOutF / BN, kTok / BM);
  gptq_gemm_wmma<<<grid, 256, 0, stream>>>(x, w16, bias, out);
}
